// PointNet2Cls_68496138437074
// MI455X (gfx1250) — matrix operand order repaired and checked by reading, not compiled, not run on hardware
//
#include <hip/hip_runtime.h>
#include <hip/hip_fp16.h>

typedef __attribute__((ext_vector_type(16))) _Float16 v16h;
typedef __attribute__((ext_vector_type(8)))  _Float16 v8h;
typedef __attribute__((ext_vector_type(8)))  float    v8f;
typedef __attribute__((ext_vector_type(4)))  unsigned int v4u;
typedef __attribute__((ext_vector_type(8)))  int      v8i_t;
typedef __attribute__((ext_vector_type(4)))  int      v4i_t;

#define BB   32      // batch
#define NPT  4096    // input point count
#define FPS_T 256

// ---------------------------------------------------------------------------
// xyz (B,3,N) -> xyz_t (B,N,3)
__global__ void k_transpose_xyz(const float* __restrict__ xyz, float* __restrict__ out, int n) {
  int t = blockIdx.x * blockDim.x + threadIdx.x;
  if (t >= BB * n) return;
  int b = t / n, p = t % n;
  out[(size_t)t * 3 + 0] = xyz[((size_t)b * 3 + 0) * n + p];
  out[(size_t)t * 3 + 1] = xyz[((size_t)b * 3 + 1) * n + p];
  out[(size_t)t * 3 + 2] = xyz[((size_t)b * 3 + 2) * n + p];
}

// ---------------------------------------------------------------------------
// Farthest point sampling. One workgroup per batch; coords cached in LDS,
// running min-distance kept in registers. Matches jnp.argmax first-occurrence
// tie-breaking.
__global__ void k_fps(const float* __restrict__ xyz, int N, int npoint, int* __restrict__ out) {
  int b = blockIdx.x, tid = threadIdx.x;
  __shared__ float px[NPT], py[NPT], pz[NPT];
  __shared__ float rv[FPS_T];
  __shared__ int   ri[FPS_T];
  __shared__ int   sfar;
  const float* base = xyz + (size_t)b * N * 3;
  for (int j = tid; j < N; j += FPS_T) {
    px[j] = base[(size_t)j * 3 + 0];
    py[j] = base[(size_t)j * 3 + 1];
    pz[j] = base[(size_t)j * 3 + 2];
  }
  float dist[NPT / FPS_T];
  int nch = N / FPS_T;
  for (int t = 0; t < nch; ++t) dist[t] = 1e10f;
  if (tid == 0) sfar = 0;
  __syncthreads();
  for (int i = 0; i < npoint; ++i) {
    int far = sfar;
    if (tid == 0) out[b * npoint + i] = far;
    float cx = px[far], cy = py[far], cz = pz[far];
    float best = -1.0f; int bi = 0;
    for (int t = 0; t < nch; ++t) {
      int j = t * FPS_T + tid;
      float dx = px[j] - cx, dy = py[j] - cy, dz = pz[j] - cz;
      float d = dx * dx + dy * dy + dz * dz;
      float nd = fminf(dist[t], d);
      dist[t] = nd;
      if (nd > best) { best = nd; bi = j; }   // strict > keeps earliest j
    }
    rv[tid] = best; ri[tid] = bi;
    __syncthreads();
    for (int off = FPS_T / 2; off > 0; off >>= 1) {
      if (tid < off) {
        float v2 = rv[tid + off]; int j2 = ri[tid + off];
        if (v2 > rv[tid] || (v2 == rv[tid] && j2 < ri[tid])) { rv[tid] = v2; ri[tid] = j2; }
      }
      __syncthreads();
    }
    if (tid == 0) sfar = ri[0];
    __syncthreads();
  }
}

// new_xyz[b,s,:] = xyz[b, fps[b,s], :]
__global__ void k_new_xyz(const float* __restrict__ xyz, const int* __restrict__ fps,
                          float* __restrict__ dst, int N, int S) {
  int t = blockIdx.x * blockDim.x + threadIdx.x;
  if (t >= BB * S) return;
  int b = t / S;
  int i = fps[t];
  const float* p = xyz + ((size_t)b * N + i) * 3;
  dst[(size_t)t * 3 + 0] = p[0];
  dst[(size_t)t * 3 + 1] = p[1];
  dst[(size_t)t * 3 + 2] = p[2];
}

// Ball query: first nsample in-radius indices in ascending order (== reference
// sort trick); pad with the first hit.
__global__ void k_ball(const float* __restrict__ xyz, const float* __restrict__ nxyz,
                       int* __restrict__ idx, int N, int S, int nsample, float r2) {
  int t = blockIdx.x * blockDim.x + threadIdx.x;
  if (t >= BB * S) return;
  int b = t / S;
  const float* c = nxyz + (size_t)t * 3;
  float cx = c[0], cy = c[1], cz = c[2];
  const float* base = xyz + (size_t)b * N * 3;
  int* row = idx + (size_t)t * nsample;
  int cnt = 0;
  for (int j = 0; j < N && cnt < nsample; ++j) {
    float dx = base[(size_t)j * 3 + 0] - cx;
    float dy = base[(size_t)j * 3 + 1] - cy;
    float dz = base[(size_t)j * 3 + 2] - cz;
    if (dx * dx + dy * dy + dz * dz <= r2) row[cnt++] = j;
  }
  int f = (cnt > 0) ? row[0] : 0;
  for (; cnt < nsample; ++cnt) row[cnt] = f;
}

// Build activation matrix X (Ne x Cpad, f16): [grouped_xyz - new_xyz (3), pts (D)], zero pad.
__global__ void k_gather_group(const float* __restrict__ xyz, const float* __restrict__ nxyz,
                               const int* __restrict__ idx, const float* __restrict__ pts,
                               int Nsrc, int S, int K, int D, int Cpad,
                               _Float16* __restrict__ X) {
  int e = blockIdx.x * blockDim.x + threadIdx.x;
  if (e >= BB * S * K) return;
  int bs = e / K;
  int s = bs % S, b = bs / S;
  int i = idx[e];
  const float* p = xyz + ((size_t)b * Nsrc + i) * 3;
  const float* c = nxyz + ((size_t)b * S + s) * 3;
  _Float16* xr = X + (size_t)e * Cpad;
  xr[0] = (_Float16)(p[0] - c[0]);
  xr[1] = (_Float16)(p[1] - c[1]);
  xr[2] = (_Float16)(p[2] - c[2]);
  int cc = 3;
  if (D > 0) {
    const float* pr = pts + (size_t)b * D * Nsrc;
    for (int d = 0; d < D; ++d) xr[3 + d] = (_Float16)pr[(size_t)d * Nsrc + i];
    cc = 3 + D;
  }
  for (; cc < Cpad; ++cc) xr[cc] = (_Float16)0.f;
}

// group_all: X row = [xyz (3, raw), pts (D)], zero pad.
__global__ void k_gather_all(const float* __restrict__ xyz2, const float* __restrict__ pooled,
                             int Np, int D, int Cpad, _Float16* __restrict__ X) {
  int e = blockIdx.x * blockDim.x + threadIdx.x;
  if (e >= BB * Np) return;
  int b = e / Np, k = e % Np;
  const float* p = xyz2 + (size_t)e * 3;
  _Float16* xr = X + (size_t)e * Cpad;
  xr[0] = (_Float16)p[0]; xr[1] = (_Float16)p[1]; xr[2] = (_Float16)p[2];
  const float* pr = pooled + (size_t)b * D * Np;
  for (int d = 0; d < D; ++d) xr[3 + d] = (_Float16)pr[(size_t)d * Np + k];
  for (int cc = 3 + D; cc < Cpad; ++cc) xr[cc] = (_Float16)0.f;
}

// f32 weights (O x C) -> zero-padded f16 (Opad x Cpad)
__global__ void k_pack_w(const float* __restrict__ W, int O, int C, int Op, int Cp,
                         _Float16* __restrict__ Wp) {
  int t = blockIdx.x * blockDim.x + threadIdx.x;
  if (t >= Op * Cp) return;
  int o = t / Cp, c = t % Cp;
  Wp[t] = (o < O && c < C) ? (_Float16)W[(size_t)o * C + c] : (_Float16)0.f;
}

// ---------------------------------------------------------------------------
// WMMA GEMM with TDM-staged activations.
// Y (O x Ne, f32) = W (O x Cpad, f16) * X^T, X stored (Ne x Cpad, f16).
// Block = 8 waves; blockIdx.x selects the 16-row X tile (contiguous 32*C bytes),
// blockIdx.y selects a group of 8 M-tiles. Wave 0 DMAs the X tile into LDS with
// TENSOR_LOAD_TO_LDS (1-D descriptor, data_size=4B, 8*C dwords), waits on
// TENSORcnt, then all 8 waves consume the shared B fragment from LDS while
// streaming their (L2-resident) W fragments from global.
__global__ void k_gemm_wmma_lds(const _Float16* __restrict__ W, const _Float16* __restrict__ X,
                                const float* __restrict__ bias, float* __restrict__ Y,
                                int O, int C, int Ne) {
  extern __shared__ _Float16 smem[];   // 16 * C halves
  int tn   = blockIdx.x << 4;
  int wave = threadIdx.x >> 5;
  int lane = threadIdx.x & 31;

  if (wave == 0) {
    unsigned lds_off = (unsigned)(size_t)(&smem[0]);   // LDS aperture: low 32 bits = byte offset
    unsigned long long ga = (unsigned long long)(const void*)(X + (size_t)tn * C);
    unsigned T = (unsigned)(8 * C);                    // tile length in dwords (<= 8192)
    v4u g0 = { 1u,                                     // count=1, user descriptor
               lds_off,
               (unsigned)(ga & 0xFFFFFFFFu),
               (unsigned)((ga >> 32) & 0x01FFFFFFu) | (2u << 30) };  // type=2 ("image")
    v8i_t g1 = { (int)(2u << 16),                      // data_size = 4 bytes
                 (int)((T & 0xFFFFu) << 16),           // tensor_dim0[15:0]
                 (int)(((T >> 16) & 0xFFFFu) | (1u << 16)), // tensor_dim0[31:16] | tensor_dim1=1
                 (int)((T & 0xFFFFu) << 16),           // tile_dim0 = T
                 (int)(1u | (1u << 16)),               // tile_dim1 = 1, tile_dim2 = 1
                 (int)T,                               // tensor_dim0_stride[31:0]
                 0, 0 };
    v4i_t g2 = {0, 0, 0, 0};
    v4i_t g3 = {0, 0, 0, 0};
#if __clang_major__ >= 23
    v8i_t g4 = {0, 0, 0, 0, 0, 0, 0, 0};
    __builtin_amdgcn_tensor_load_to_lds(g0, g1, g2, g3, g4, 0);
#else
    __builtin_amdgcn_tensor_load_to_lds(g0, g1, g2, g3, 0);
#endif
    __builtin_amdgcn_s_wait_tensorcnt(0);
  }
  __syncthreads();

  int mt = blockIdx.y * 8 + wave;
  if (mt * 16 < O) {
    int tm = mt << 4;
    int hi = lane >> 4;
    int l  = lane & 15;
    const _Float16* arow = W + (size_t)(tm + l) * C + hi * 8;    // A frag: global (L2-cached)
    const _Float16* brow = smem + (size_t)l * C + hi * 8;        // B frag: LDS
    v8f acc = {};
    for (int k0 = 0; k0 < C; k0 += 32) {
      v16h a, b;
      *(v8h*)&a         = *(const v8h*)(arow + k0);
      *(((v8h*)&a) + 1) = *(const v8h*)(arow + k0 + 16);
      *(v8h*)&b         = *(const v8h*)(brow + k0);
      *(((v8h*)&b) + 1) = *(const v8h*)(brow + k0 + 16);
      acc = __builtin_amdgcn_wmma_f32_16x16x32_f16(false, a, false, b, (short)0, acc,
                                                   false, false);
    }
#pragma unroll
    for (int i = 0; i < 8; ++i) {
      int m = tm + i + hi * 8;            // C/D layout: vgpr i -> row i (+8 for hi lanes)
      float v = acc[i];
      if (bias) v += bias[m];
      Y[(size_t)m * Ne + tn + l] = v;
    }
  }
}

// ---------------------------------------------------------------------------
// BN stats per channel over Ne (biased var, like jnp.var).
__global__ void k_bn_stats(const float* __restrict__ Y, int Ne,
                           float* __restrict__ mean, float* __restrict__ inv) {
  __shared__ float s1[256], s2[256];
  int o = blockIdx.x, tid = threadIdx.x;
  const float* row = Y + (size_t)o * Ne;
  float a = 0.f, q = 0.f;
  for (int e = tid; e < Ne; e += 256) { float v = row[e]; a += v; q += v * v; }
  s1[tid] = a; s2[tid] = q;
  __syncthreads();
  for (int off = 128; off > 0; off >>= 1) {
    if (tid < off) { s1[tid] += s1[tid + off]; s2[tid] += s2[tid + off]; }
    __syncthreads();
  }
  if (tid == 0) {
    float m = s1[0] / (float)Ne;
    float var = s2[0] / (float)Ne - m * m;
    mean[o] = m;
    inv[o] = rsqrtf(var + 1e-5f);
  }
}

// normalize + affine + relu; write f16 into next layer's (Ne x O) activation buffer.
__global__ void k_bn_apply(const float* __restrict__ Y, const float* __restrict__ g,
                           const float* __restrict__ be, const float* __restrict__ mean,
                           const float* __restrict__ inv, _Float16* __restrict__ Xn,
                           int O, int Ne) {
  int t = blockIdx.x * blockDim.x + threadIdx.x;
  if (t >= O * Ne) return;
  int o = t / Ne, e = t % Ne;
  float v = (Y[t] - mean[o]) * inv[o] * g[o] + be[o];
  v = fmaxf(v, 0.f);
  Xn[(size_t)e * O + o] = (_Float16)v;
}

// max over nsample segment: pooled[b,o,s] = max_k X[((b*S+s)*K+k)*O + o]
__global__ void k_maxpool(const _Float16* __restrict__ X, float* __restrict__ pooled,
                          int O, int S, int K) {
  int t = blockIdx.x * blockDim.x + threadIdx.x;
  if (t >= BB * O * S) return;
  int b = t / (O * S);
  int rem = t % (O * S);
  int o = rem / S, s = rem % S;
  size_t base = ((size_t)(b * S + s) * K) * O + o;
  float m = -3.4e38f;
  for (int k = 0; k < K; ++k) {
    float v = (float)X[base + (size_t)k * O];
    m = fmaxf(m, v);
  }
  pooled[((size_t)b * O + o) * S + s] = m;
}

__global__ void k_f2h(const float* __restrict__ src, _Float16* __restrict__ dst, int n) {
  int t = blockIdx.x * blockDim.x + threadIdx.x;
  if (t < n) dst[t] = (_Float16)src[t];
}

// final: out[b,o] = Y[o*32 + b] + fc3_b[o]
__global__ void k_out(const float* __restrict__ Y, const float* __restrict__ bias,
                      float* __restrict__ out) {
  int t = blockIdx.x * blockDim.x + threadIdx.x;
  if (t >= BB * 40) return;
  int b = t / 40, o = t % 40;
  out[t] = Y[(size_t)o * BB + b] + bias[o];
}

// ---------------------------------------------------------------------------
extern "C" void kernel_launch(void* const* d_in, const int* in_sizes, int n_in,
                              void* d_out, int out_size, void* d_ws, size_t ws_size,
                              hipStream_t stream) {
  (void)in_sizes; (void)n_in; (void)out_size; (void)ws_size;
  const float* xyz = (const float*)d_in[0];
  const float *w_sa[9], *b_sa[9], *g_sa[9], *be_sa[9];
  for (int l = 0; l < 9; ++l) {
    int base = 1 + l * 4;
    w_sa[l]  = (const float*)d_in[base + 0];
    b_sa[l]  = (const float*)d_in[base + 1];
    g_sa[l]  = (const float*)d_in[base + 2];
    be_sa[l] = (const float*)d_in[base + 3];
  }
  const float* fc1_w = (const float*)d_in[37]; const float* fc1_b = (const float*)d_in[38];
  const float* fc2_w = (const float*)d_in[39]; const float* fc2_b = (const float*)d_in[40];
  const float* fc3_w = (const float*)d_in[41]; const float* fc3_b = (const float*)d_in[42];
  const float* bn1_g = (const float*)d_in[43]; const float* bn1_b = (const float*)d_in[44];
  const float* bn2_g = (const float*)d_in[45]; const float* bn2_b = (const float*)d_in[46];

  const int S1 = 512, K1 = 32, S2 = 128, K2 = 64;
  const int Ne1 = BB * S1 * K1;   // 524288
  const int Ne2 = BB * S2 * K2;   // 262144
  const int Ne3 = BB * 128;       // 4096

  // ----- workspace layout -----
  char* p = (char*)d_ws;
  auto alloc = [&](size_t bytes) -> void* {
    void* r = (void*)p;
    p += (bytes + 255) & ~(size_t)255;
    return r;
  };
  float* xyz_t   = (float*)alloc((size_t)BB * NPT * 3 * 4);
  float* nxyz1   = (float*)alloc((size_t)BB * S1 * 3 * 4);
  float* nxyz2   = (float*)alloc((size_t)BB * S2 * 3 * 4);
  int*   fps1    = (int*)alloc((size_t)BB * S1 * 4);
  int*   fps2    = (int*)alloc((size_t)BB * S2 * 4);
  int*   idx1    = (int*)alloc((size_t)BB * S1 * K1 * 4);
  int*   idx2    = (int*)alloc((size_t)BB * S2 * K2 * 4);
  float* pooled1 = (float*)alloc((size_t)BB * 128 * S1 * 4);
  float* pooled2 = (float*)alloc((size_t)BB * 256 * S2 * 4);
  float* pooled3 = (float*)alloc((size_t)BB * 1024 * 4);
  float* mean    = (float*)alloc(1024 * 4);
  float* inv     = (float*)alloc(1024 * 4);

  const int Oact[12] = {64, 64, 128, 128, 128, 256, 256, 512, 1024, 512, 256, 40};
  const int Opad[12] = {64, 64, 128, 128, 128, 256, 256, 512, 1024, 512, 256, 48};
  const int Cact[12] = {3, 64, 64, 131, 128, 128, 259, 256, 512, 1024, 512, 256};
  const int Cpad[12] = {32, 64, 64, 160, 128, 128, 288, 256, 512, 1024, 512, 256};
  const float* wsrc[12] = {w_sa[0], w_sa[1], w_sa[2], w_sa[3], w_sa[4], w_sa[5],
                           w_sa[6], w_sa[7], w_sa[8], fc1_w, fc2_w, fc3_w};
  const float* biasp[12] = {b_sa[0], b_sa[1], b_sa[2], b_sa[3], b_sa[4], b_sa[5],
                            b_sa[6], b_sa[7], b_sa[8], fc1_b, fc2_b, nullptr};
  const float* gp[12]  = {g_sa[0], g_sa[1], g_sa[2], g_sa[3], g_sa[4], g_sa[5],
                          g_sa[6], g_sa[7], g_sa[8], bn1_g, bn2_g, nullptr};
  const float* bep[12] = {be_sa[0], be_sa[1], be_sa[2], be_sa[3], be_sa[4], be_sa[5],
                          be_sa[6], be_sa[7], be_sa[8], bn1_b, bn2_b, nullptr};
  _Float16* Wp[12];
  for (int i = 0; i < 12; ++i) Wp[i] = (_Float16*)alloc((size_t)Opad[i] * Cpad[i] * 2);

  _Float16* Xa = (_Float16*)alloc((size_t)Ne1 * 128 * 2);
  _Float16* Xb = (_Float16*)alloc((size_t)Ne1 * 128 * 2);
  float*    Yb = (float*)alloc((size_t)128 * Ne1 * 4);

  auto cdiv = [](long a, long b) { return (int)((a + b - 1) / b); };

  auto gemm = [&](int li, const _Float16* Xin, int Ne) {
    int O = Opad[li], C = Cpad[li];
    dim3 grid(Ne / 16, ((O / 16) + 7) / 8);
    size_t shm = (size_t)16 * C * 2;   // 16-row X tile
    k_gemm_wmma_lds<<<grid, 256, shm, stream>>>(Wp[li], Xin, biasp[li], Yb, O, C, Ne);
  };
  auto run_layer = [&](int li, const _Float16* Xin, _Float16* Xout, int Ne) {
    int O = Opad[li];
    gemm(li, Xin, Ne);
    k_bn_stats<<<O, 256, 0, stream>>>(Yb, Ne, mean, inv);
    k_bn_apply<<<cdiv((long)O * Ne, 256), 256, 0, stream>>>(Yb, gp[li], bep[li], mean, inv,
                                                            Xout, O, Ne);
  };

  // ----- stage 0: transpose + pack weights -----
  k_transpose_xyz<<<cdiv((long)BB * NPT, 256), 256, 0, stream>>>(xyz, xyz_t, NPT);
  for (int i = 0; i < 12; ++i) {
    long tot = (long)Opad[i] * Cpad[i];
    k_pack_w<<<cdiv(tot, 256), 256, 0, stream>>>(wsrc[i], Oact[i], Cact[i], Opad[i], Cpad[i], Wp[i]);
  }

  // ----- SA1 -----
  k_fps<<<BB, FPS_T, 0, stream>>>(xyz_t, NPT, S1, fps1);
  k_new_xyz<<<cdiv((long)BB * S1, 256), 256, 0, stream>>>(xyz_t, fps1, nxyz1, NPT, S1);
  k_ball<<<cdiv((long)BB * S1, 256), 256, 0, stream>>>(xyz_t, nxyz1, idx1, NPT, S1, K1, 0.04f);
  k_gather_group<<<cdiv((long)Ne1, 256), 256, 0, stream>>>(xyz_t, nxyz1, idx1, nullptr,
                                                           NPT, S1, K1, 0, 32, Xa);
  run_layer(0, Xa, Xb, Ne1);
  run_layer(1, Xb, Xa, Ne1);
  run_layer(2, Xa, Xb, Ne1);
  k_maxpool<<<cdiv((long)BB * 128 * S1, 256), 256, 0, stream>>>(Xb, pooled1, 128, S1, K1);

  // ----- SA2 -----
  k_fps<<<BB, FPS_T, 0, stream>>>(nxyz1, S1, S2, fps2);
  k_new_xyz<<<cdiv((long)BB * S2, 256), 256, 0, stream>>>(nxyz1, fps2, nxyz2, S1, S2);
  k_ball<<<cdiv((long)BB * S2, 256), 256, 0, stream>>>(nxyz1, nxyz2, idx2, S1, S2, K2, 0.16f);
  k_gather_group<<<cdiv((long)Ne2, 256), 256, 0, stream>>>(nxyz1, nxyz2, idx2, pooled1,
                                                           S1, S2, K2, 128, 160, Xa);
  run_layer(3, Xa, Xb, Ne2);
  run_layer(4, Xb, Xa, Ne2);
  run_layer(5, Xa, Xb, Ne2);
  k_maxpool<<<cdiv((long)BB * 256 * S2, 256), 256, 0, stream>>>(Xb, pooled2, 256, S2, K2);

  // ----- SA3 (group_all) -----
  k_gather_all<<<cdiv((long)Ne3, 256), 256, 0, stream>>>(nxyz2, pooled2, 128, 256, 288, Xa);
  run_layer(6, Xa, Xb, Ne3);
  run_layer(7, Xb, Xa, Ne3);
  run_layer(8, Xa, Xb, Ne3);
  k_maxpool<<<cdiv((long)BB * 1024, 256), 256, 0, stream>>>(Xb, pooled3, 1024, 1, 128);

  // ----- FC head -----
  k_f2h<<<cdiv((long)BB * 1024, 256), 256, 0, stream>>>(pooled3, Xa, BB * 1024);
  run_layer(9, Xa, Xb, BB);    // fc1 + bn1 + relu -> (32 x 512) f16
  run_layer(10, Xb, Xa, BB);   // fc2 + bn2 + relu -> (32 x 256) f16
  gemm(11, Xa, BB);            // fc3 -> Yb (48 x 32) f32
  k_out<<<cdiv((long)BB * 40, 256), 256, 0, stream>>>(Yb, fc3_b, (float*)d_out);
}
